// SUnet_15058155340345
// MI455X (gfx1250) — matrix operand order repaired and checked by reading, not compiled, not run on hardware
//
#include <hip/hip_runtime.h>

// ---------------------------------------------------------------------------
// Shifted-window attention (Swin block) fused kernel for gfx1250 (MI455X).
// One workgroup (8 wave32) per (batch, window). All intermediates in LDS.
// Matrix math on v_wmma_f32_16x16x32_f16; weight staging via
// global_load_async_to_lds_b128 (ASYNCcnt).
// ---------------------------------------------------------------------------

typedef _Float16 half_t;
typedef __attribute__((ext_vector_type(16))) _Float16 v16h;
typedef __attribute__((ext_vector_type(8)))  _Float16 v8h;
typedef __attribute__((ext_vector_type(8)))  float    v8f;

#define WSZ      7
#define DISP     3
#define HEADS    4
#define HEAD_DIM 32
#define DIM      128
#define HDIM     56
#define WDIM     56
#define NWH      8
#define NWW      8
#define NTOK     49        // 7*7 tokens per window
#define MPAD     64        // padded token count (4 x 16 WMMA tiles)
#define BATCH    16

// ---- WMMA fragment loaders (wave32, CDNA5 layouts) -------------------------
// A (16x32 f16, row-major source, stride ld):
//   lanes 0-15 : M=lane,    K = k0+[0..7]  then k0+[16..23]
//   lanes 16-31: M=lane-16, K = k0+[8..15] then k0+[24..31]
__device__ __forceinline__ v16h frag_a(const half_t* base, int ld, int m0, int k0) {
  int lane = threadIdx.x & 31;
  int m  = m0 + (lane & 15);
  int kb = k0 + ((lane >> 4) << 3);
  const half_t* p = base + m * ld + kb;
  v8h lo = *(const v8h*)(p);        // K = kb .. kb+7
  v8h hi = *(const v8h*)(p + 16);   // K = kb+16 .. kb+23
  v16h f;
#pragma unroll
  for (int i = 0; i < 8; ++i) { f[i] = lo[i]; f[i + 8] = hi[i]; }
  return f;
}

// B (32x16 f16) loaded from a TRANSPOSED (NxK row-major) operand, stride ld:
//   lane holds column N = n0 + lane%16, K = k0 + 8*(lane/16) + [0..7] then k0 + 16 + 8*(lane/16) + [0..7]
__device__ __forceinline__ v16h frag_b(const half_t* baseT, int ld, int n0, int k0) {
  int lane = threadIdx.x & 31;
  int n  = n0 + (lane & 15);
  int kb = k0 + ((lane >> 4) << 3);
  const half_t* p = baseT + n * ld + kb;
  v8h lo = *(const v8h*)(p);
  v8h hi = *(const v8h*)(p + 16);
  v16h f;
#pragma unroll
  for (int i = 0; i < 8; ++i) { f[i] = lo[i]; f[i + 8] = hi[i]; }
  return f;
}

__device__ __forceinline__ v8f wmma_f16(v16h a, v16h b, v8f c) {
  return __builtin_amdgcn_wmma_f32_16x16x32_f16(false, a, false, b, (short)0, c,
                                                false, false);
}

// ---- Async global->LDS DMA (gfx1250, ASYNCcnt-tracked) ---------------------
// Per-lane 16B copy: LDS[vdst] = MEM[vaddr]. LDS byte address is the low 32
// bits of the generic pointer (LDS aperture keeps the offset in addr[31:0]).
__device__ __forceinline__ void async_copy_b128(void* lds_dst, const void* gsrc) {
  unsigned int l = (unsigned int)(unsigned long long)lds_dst;
  unsigned long long g = (unsigned long long)gsrc;
  asm volatile("global_load_async_to_lds_b128 %0, %1, off"
               :: "v"(l), "v"(g) : "memory");
}
__device__ __forceinline__ void async_wait0() {
  asm volatile("s_wait_asynccnt 0" ::: "memory");
}

// ---- Kernel 1: transpose + f16-convert weights into workspace --------------
__global__ void swin_prep_weights(const float* __restrict__ w_qkv,
                                  const float* __restrict__ w_out,
                                  half_t* __restrict__ wqkv_t,   // [384][128]
                                  half_t* __restrict__ wout_t) { // [128][128]
  int stride = gridDim.x * blockDim.x;
  int i0 = blockIdx.x * blockDim.x + threadIdx.x;
  for (int idx = i0; idx < 3 * DIM * DIM; idx += stride) {
    int n = idx >> 7, k = idx & 127;
    wqkv_t[idx] = (half_t)w_qkv[k * (3 * DIM) + n];
  }
  for (int idx = i0; idx < DIM * DIM; idx += stride) {
    int n = idx >> 7, k = idx & 127;
    wout_t[idx] = (half_t)w_out[k * DIM + n];
  }
}

// ---- Kernel 2: fused per-window attention ----------------------------------
__global__ __launch_bounds__(256)
void swin_window_attn(const float* __restrict__ x,
                      const half_t* __restrict__ wqkv_t,
                      const float* __restrict__ pos_emb,
                      const half_t* __restrict__ wout_t,
                      const float* __restrict__ b_out,
                      float* __restrict__ out) {
  // LDS layout (90,880 B; gfx1250 WGP has 320KB):
  //  [0      ,16K): sX  64x128 f16  -> later sDots 64x64 f32
  //  [16K    ,32K): weight chunk staging -> later sO 64x128 f16
  //  [32K    ,48K): sQ  64x128 f16
  //  [48K    ,64K): sK  64x128 f16
  //  [64K    ,80K): sVt 128x64 f16 (V transposed: [dim][token])
  //  [80K    ,88K): sS  attn 64x64 f16 / w_out chunk 32x128 f16
  //  [88K    ,+676): sPE 13x13 f32
  __shared__ __align__(16) unsigned char smem_raw[90880];
  half_t* sX    = (half_t*)(smem_raw);
  half_t* sO    = (half_t*)(smem_raw + 16384);
  half_t* sQ    = (half_t*)(smem_raw + 32768);
  half_t* sK    = (half_t*)(smem_raw + 49152);
  half_t* sVt   = (half_t*)(smem_raw + 65536);
  half_t* sS    = (half_t*)(smem_raw + 81920);
  float*  sPE   = (float*)(smem_raw + 90112);
  float*  sDots = (float*)(smem_raw);   // aliases sX (dead after QKV)

  const int tid  = threadIdx.x;
  const int lane = tid & 31;
  const int wave = tid >> 5;

  const int blk = blockIdx.x;
  const int b   = blk >> 6;
  const int w   = blk & 63;
  const int wh  = w >> 3;
  const int ww  = w & 7;
  const bool maskUL = (wh == NWH - 1);   // bottom window row: shift in H
  const bool maskLR = (ww == NWW - 1);   // right window col: shift in W

  if (tid < 169) sPE[tid] = pos_emb[tid];

  // Stage rolled x window as f16; rows 49..63 zeroed (WMMA padding).
  {
    int r  = tid >> 2;            // 0..63
    int c0 = (tid & 3) * 32;
    if (r < NTOK) {
      int gr = (wh * WSZ + r / WSZ + DISP) % HDIM;
      int gc = (ww * WSZ + r % WSZ + DISP) % WDIM;
      const float* src = x + (((size_t)b * HDIM + gr) * WDIM + gc) * DIM + c0;
#pragma unroll
      for (int i = 0; i < 32; i += 4) {
        float4 f = *(const float4*)(src + i);
        sX[r * DIM + c0 + i + 0] = (half_t)f.x;
        sX[r * DIM + c0 + i + 1] = (half_t)f.y;
        sX[r * DIM + c0 + i + 2] = (half_t)f.z;
        sX[r * DIM + c0 + i + 3] = (half_t)f.w;
      }
    } else {
#pragma unroll
      for (int i = 0; i < 32; ++i) sX[r * DIM + c0 + i] = (half_t)0.f;
    }
  }

  // ---- QKV GEMM: (64x128) @ (128x384), 6 chunks of 64 output columns ------
  for (int chunk = 0; chunk < 6; ++chunk) {
    __syncthreads();
    {  // async-DMA 64x128 f16 transposed-weight rows into sO region
      int row = tid >> 2;
      int c0  = (tid & 3) * 32;
      const half_t* src = wqkv_t + ((size_t)(chunk * 64 + row)) * DIM + c0;
      half_t* dst = sO + row * DIM + c0;
#pragma unroll
      for (int i = 0; i < 4; ++i) async_copy_b128(dst + i * 8, src + i * 8);
      async_wait0();
    }
    __syncthreads();
    for (int t = wave; t < 16; t += 8) {   // 4 mt x 4 nt tiles
      int mt = t >> 2, ntc = t & 3;
      v8f acc = {};
#pragma unroll
      for (int ks = 0; ks < 4; ++ks) {
        v16h a  = frag_a(sX, DIM, mt * 16, ks * 32);
        v16h bf = frag_b(sO, DIM, ntc * 16, ks * 32);
        acc = wmma_f16(a, bf, acc);
      }
      // Tile base column is wave-uniform -> uniform Q/K/V routing.
      int ncb  = chunk * 64 + ntc * 16;     // uniform
      int nloc = lane & 15;
      int mb   = mt * 16 + ((lane >> 4) << 3);
      if (ncb < 128) {                       // -> Q, row-major
        half_t* dst = sQ + ncb + nloc;
#pragma unroll
        for (int r = 0; r < 8; ++r) dst[(mb + r) * DIM] = (half_t)acc[r];
      } else if (ncb < 256) {                // -> K, row-major
        half_t* dst = sK + (ncb - 128) + nloc;
#pragma unroll
        for (int r = 0; r < 8; ++r) dst[(mb + r) * DIM] = (half_t)acc[r];
      } else {                               // -> V transposed: 8 consecutive m
        v8h pk;
#pragma unroll
        for (int r = 0; r < 8; ++r) pk[r] = (half_t)acc[r];
        *(v8h*)(sVt + (ncb - 256 + nloc) * MPAD + mb) = pk;  // ds_store_b128
      }
    }
  }

  // ---- Attention, one head at a time --------------------------------------
  const float scale = 0.17677669529663687f;  // 1/sqrt(32)
  for (int h = 0; h < HEADS; ++h) {
    __syncthreads();
    // dots = Q_h * K_h^T  (64x64, K=32 -> one WMMA per 16x16 tile)
    for (int t = wave; t < 16; t += 8) {
      int mt = t >> 2, nt = t & 3;
      v16h a  = frag_a(sQ, DIM, mt * 16, h * HEAD_DIM);
      v16h bf = frag_b(sK, DIM, nt * 16, h * HEAD_DIM);
      v8f acc = {};
      acc = wmma_f16(a, bf, acc);
      int n  = nt * 16 + (lane & 15);
      int mb = mt * 16 + ((lane >> 4) << 3);
#pragma unroll
      for (int r = 0; r < 8; ++r) sDots[(mb + r) * MPAD + n] = acc[r];
    }
    __syncthreads();
    // scale + rel-pos bias + shift mask + softmax -> f16 attn in sS
    if (tid < MPAD) {
      int i = tid;
      if (i < NTOK) {
        int xi = i / WSZ, yi = i % WSZ;
        float mx = -3.0e38f;
        for (int j = 0; j < NTOK; ++j) {
          int xj = j / WSZ, yj = j % WSZ;
          float d = sDots[i * MPAD + j] * scale +
                    sPE[(xj - xi + WSZ - 1) * 13 + (yj - yi + WSZ - 1)];
          bool msk = (maskUL && ((xi >= 4) != (xj >= 4))) ||
                     (maskLR && ((yi >= 4) != (yj >= 4)));
          d = msk ? -1.0e30f : d;
          sDots[i * MPAD + j] = d;
          mx = fmaxf(mx, d);
        }
        float sum = 0.f;
        for (int j = 0; j < NTOK; ++j) {
          float e = __expf(sDots[i * MPAD + j] - mx);
          sDots[i * MPAD + j] = e;
          sum += e;
        }
        float inv = 1.f / sum;
        for (int j = 0; j < NTOK; ++j)
          sS[i * MPAD + j] = (half_t)(sDots[i * MPAD + j] * inv);
        for (int j = NTOK; j < MPAD; ++j) sS[i * MPAD + j] = (half_t)0.f;
      } else {
        for (int j = 0; j < MPAD; ++j) sS[i * MPAD + j] = (half_t)0.f;
      }
    }
    __syncthreads();
    // O_h = attn * V_h  (64x32, K=64) -> columns [h*32, h*32+32) of sO
    {
      int mt = wave >> 1, nt = wave & 1;   // 8 tiles, one per wave
      v8f acc = {};
#pragma unroll
      for (int ks = 0; ks < 2; ++ks) {
        v16h a  = frag_a(sS, MPAD, mt * 16, ks * 32);
        v16h bf = frag_b(sVt, MPAD, h * HEAD_DIM + nt * 16, ks * 32);
        acc = wmma_f16(a, bf, acc);
      }
      int n  = h * HEAD_DIM + nt * 16 + (lane & 15);
      int mb = mt * 16 + ((lane >> 4) << 3);
#pragma unroll
      for (int r = 0; r < 8; ++r) sO[(mb + r) * DIM + n] = (half_t)acc[r];
    }
  }

  // ---- Output projection: (64x128) @ (128x128) + bias, rolled store -------
  for (int chunk = 0; chunk < 4; ++chunk) {   // 32 output columns per chunk
    __syncthreads();
    {  // async-DMA w_out^T rows [chunk*32, +32) x 128 into sS
      int row = tid >> 3;            // 0..31
      int c0  = (tid & 7) * 16;
      const half_t* src = wout_t + ((size_t)(chunk * 32 + row)) * DIM + c0;
      half_t* dst = sS + row * DIM + c0;
      async_copy_b128(dst,     src);
      async_copy_b128(dst + 8, src + 8);
      async_wait0();
    }
    __syncthreads();
    {
      int mt = wave >> 1, nt = wave & 1;   // 8 tiles, one per wave
      v8f acc = {};
#pragma unroll
      for (int ks = 0; ks < 4; ++ks) {
        v16h a  = frag_a(sO, DIM, mt * 16, ks * 32);
        v16h bf = frag_b(sS, DIM, nt * 16, ks * 32);
        acc = wmma_f16(a, bf, acc);
      }
      int ncol = chunk * 32 + nt * 16 + (lane & 15);
      float bb = b_out[ncol];
      int mb = mt * 16 + ((lane >> 4) << 3);
#pragma unroll
      for (int r = 0; r < 8; ++r) {
        int m = mb + r;
        if (m < NTOK) {
          int gr = (wh * WSZ + m / WSZ + DISP) % HDIM;
          int gc = (ww * WSZ + m % WSZ + DISP) % WDIM;
          out[(((size_t)b * HDIM + gr) * WDIM + gc) * DIM + ncol] = acc[r] + bb;
        }
      }
    }
  }
}

// ---------------------------------------------------------------------------
extern "C" void kernel_launch(void* const* d_in, const int* in_sizes, int n_in,
                              void* d_out, int out_size, void* d_ws, size_t ws_size,
                              hipStream_t stream) {
  (void)in_sizes; (void)n_in; (void)out_size; (void)ws_size;
  const float* x       = (const float*)d_in[0];
  const float* w_qkv   = (const float*)d_in[1];
  const float* pos_emb = (const float*)d_in[2];
  const float* w_out   = (const float*)d_in[3];
  const float* b_out   = (const float*)d_in[4];
  float* out = (float*)d_out;

  half_t* wqkv_t = (half_t*)d_ws;                                    // 96 KB
  half_t* wout_t = (half_t*)((char*)d_ws + (size_t)3 * DIM * DIM * sizeof(half_t)); // 32 KB

  swin_prep_weights<<<64, 256, 0, stream>>>(w_qkv, w_out, wqkv_t, wout_t);
  swin_window_attn<<<BATCH * NWH * NWW, 256, 0, stream>>>(x, wqkv_t, pos_emb,
                                                          wout_t, b_out, out);
}
